// Attention3D_85126251807389
// MI455X (gfx1250) — matrix operand order repaired and checked by reading, not compiled, not run on hardware
//
#include <hip/hip_runtime.h>

typedef __attribute__((ext_vector_type(16))) _Float16 v16h;
typedef __attribute__((ext_vector_type(8)))  _Float16 v8h;
typedef __attribute__((ext_vector_type(8)))  float    v8f;
typedef __attribute__((ext_vector_type(4)))  float    v4f;

static constexpr int BATCH  = 2;
static constexpr int CH     = 256;
static constexpr int NSP    = 4096;  // 16*16*16 spatial positions
static constexpr int NHEADS = 4;
static constexpr int DHEAD  = 64;

union AFrag { v16h v; v8h h[2]; };

__device__ __forceinline__ v8f wmma_f16f32(v16h a, v16h b, v8f c) {
  // D = A(16x32 f16) * B(32x16 f16) + C(16x16 f32)
  return __builtin_amdgcn_wmma_f32_16x16x32_f16(false, a, false, b, (short)0, c,
                                                false, false);
}

// Wave-private LDS ordering: DS ops from one wave complete in order (ISA
// "LDS: in order"), lanes are lockstep in wave32; we only need to stop the
// compiler from reordering the transpose load above the store. No hardware
// barrier required.
__device__ __forceinline__ void wave_lds_fence() {
  __builtin_amdgcn_fence(__ATOMIC_RELEASE, "wavefront");
  __builtin_amdgcn_wave_barrier();
  __builtin_amdgcn_fence(__ATOMIC_ACQUIRE, "wavefront");
}

__device__ __forceinline__ float redmax16(float v) {
  v = fmaxf(v, __shfl_xor(v, 1, 32));
  v = fmaxf(v, __shfl_xor(v, 2, 32));
  v = fmaxf(v, __shfl_xor(v, 4, 32));
  v = fmaxf(v, __shfl_xor(v, 8, 32));
  return v;
}
__device__ __forceinline__ float redsum16(float v) {
  v += __shfl_xor(v, 1, 32);
  v += __shfl_xor(v, 2, 32);
  v += __shfl_xor(v, 4, 32);
  v += __shfl_xor(v, 8, 32);
  return v;
}

// ---------------------------------------------------------------------------
// Kernel 0: convert f32 weights to f16 staging buffers.
// ---------------------------------------------------------------------------
__global__ void cvt_w_kernel(const float* __restrict__ qw, const float* __restrict__ pw,
                             _Float16* __restrict__ wq16, _Float16* __restrict__ wp16) {
  const int stride = gridDim.x * blockDim.x;
  const int i0 = blockIdx.x * blockDim.x + threadIdx.x;
  for (int i = i0; i < 3 * CH * CH; i += stride) wq16[i] = (_Float16)qw[i];
  for (int i = i0; i < CH * CH;     i += stride) wp16[i] = (_Float16)pw[i];
}

// ---------------------------------------------------------------------------
// Kernel 1: GroupNorm over [b, 256, 4096], 8 groups -> writes xn transposed
// as f16 [b][n][c] (the QKV GEMM B-operand layout).
// One block per (b, group); group region is a contiguous 32*4096 f32 span.
// ---------------------------------------------------------------------------
__global__ void gn_kernel(const float* __restrict__ x, const float* __restrict__ gw,
                          const float* __restrict__ gb, _Float16* __restrict__ xnt) {
  const int blk  = blockIdx.x;          // 16 blocks
  const int bidx = blk >> 3, g = blk & 7;
  const int tid  = threadIdx.x;         // 256 threads
  const float* xg = x + (size_t)(bidx * CH + g * 32) * NSP;

  float s = 0.f, ss = 0.f;
  for (int i = tid; i < 32 * NSP; i += 256) { float v = xg[i]; s += v; ss += v * v; }
  for (int m = 1; m < 32; m <<= 1) { s += __shfl_xor(s, m, 32); ss += __shfl_xor(ss, m, 32); }

  __shared__ float red[2][8];
  const int wv = tid >> 5, lane = tid & 31;
  if (lane == 0) { red[0][wv] = s; red[1][wv] = ss; }
  __syncthreads();
  float S = 0.f, SS = 0.f;
#pragma unroll
  for (int j = 0; j < 8; ++j) { S += red[0][j]; SS += red[1][j]; }
  const float inv_n = 1.0f / (32.0f * NSP);
  const float mean  = S * inv_n;
  const float var   = SS * inv_n - mean * mean;
  const float rstd  = rsqrtf(var + 1e-5f);

  for (int i = tid; i < 32 * NSP; i += 256) {
    const int ch = g * 32 + (i >> 12);
    const int sp = i & (NSP - 1);
    const float y = (xg[i] - mean) * rstd * gw[ch] + gb[ch];
    xnt[((size_t)bidx * NSP + sp) * CH + ch] = (_Float16)y;
  }
}

// ---------------------------------------------------------------------------
// Kernel 2: QKV GEMM  qkv[b, 768, n] = qkv_w[768,256] @ xn[b,256,n] + bias.
// Each wave computes a 16(M) x 64(N) strip with 32 WMMAs (K=256 in 8 steps).
// q,k written TRANSPOSED [n][256] (packed 16B stores), v written [c][n].
// ---------------------------------------------------------------------------
__global__ void __launch_bounds__(128)
qkv_kernel(const _Float16* __restrict__ wq16, const float* __restrict__ qkv_b,
           const _Float16* __restrict__ xnt, _Float16* __restrict__ qt,
           _Float16* __restrict__ ktb, _Float16* __restrict__ vbf) {
  const int lane = threadIdx.x & 31;
  const int wv   = threadIdx.x >> 5;
  const int task = blockIdx.x * 4 + wv;        // 6144 tasks
  const int bidx = task / 3072;
  const int rem  = task % 3072;
  const int Mb   = (rem / 64) * 16;            // output channel base 0..752
  const int Nb   = (rem % 64) * 64;            // spatial base
  const int ln = lane & 15, hl = lane >> 4;

  v8f acc[4] = {};
  const _Float16* arow = wq16 + (size_t)(Mb + ln) * CH;
  const _Float16* brow = xnt + ((size_t)bidx * NSP + Nb) * CH;

#pragma unroll
  for (int kk = 0; kk < 8; ++kk) {
    const int kb = kk * 32;
    AFrag a;
    a.h[0] = *(const v8h*)(arow + kb + 16 * hl);
    a.h[1] = *(const v8h*)(arow + kb + 16 * hl + 8);
#pragma unroll
    for (int t = 0; t < 4; ++t) {
      v16h bv = *(const v16h*)(brow + (size_t)(t * 16 + ln) * CH + kb + 16 * hl);
      acc[t] = wmma_f16f32(a.v, bv, acc[t]);
    }
  }

  const v8f bias = *(const v8f*)(qkv_b + Mb + 8 * hl);
  const int sec = Mb >> 8;                     // 0=q, 1=k, 2=v
  const int mc0 = Mb & 255;
#pragma unroll
  for (int t = 0; t < 4; ++t) {
    const int sp = Nb + t * 16 + ln;
    v8f r = acc[t] + bias;
    if (sec == 2) {                            // v: channel-major [c][n]
      const int chb = mc0 + 8 * hl;
#pragma unroll
      for (int rr = 0; rr < 8; ++rr)
        vbf[(size_t)bidx * CH * NSP + (size_t)(chb + rr) * NSP + sp] = (_Float16)r[rr];
    } else {                                   // q/k: transposed [n][c], packed 16B
      _Float16* dst = (sec == 0 ? qt : ktb) +
                      ((size_t)bidx * NSP + sp) * CH + mc0 + 8 * hl;
      v8h pk;
#pragma unroll
      for (int rr = 0; rr < 8; ++rr) pk[rr] = (_Float16)r[rr];
      *(v8h*)dst = pk;
    }
  }
}

// ---------------------------------------------------------------------------
// Kernel 3: flash attention per (b, head). Block = 4 waves = 64 queries,
// each wave owns 16 queries x full dh=64 accumulator (4 f32 C-tiles).
// Streams 128 key tiles of 32: 4 WMMAs for S, online softmax (shfl_xor row
// reductions), P transposed C->A via 1KB/wave LDS (wave-fence only, no
// block barrier), 4 WMMAs for P*V with V fragments preloaded above the
// softmax so the exp chain hides their latency.
// Output written to at[b][n][256] f16.
// ---------------------------------------------------------------------------
__global__ void __launch_bounds__(128)
attn_kernel(const _Float16* __restrict__ qt, const _Float16* __restrict__ ktb,
            const _Float16* __restrict__ vbf, _Float16* __restrict__ at) {
  __shared__ __align__(32) _Float16 pLds[4][16][32];
  const int lane = threadIdx.x & 31;
  const int wv   = threadIdx.x >> 5;
  const int ln = lane & 15, hl = lane >> 4;
  const int bh    = blockIdx.x >> 6;           // 0..7
  const int qt64  = blockIdx.x & 63;
  const int bidx  = bh >> 2, h = bh & 3;
  const int qbase = qt64 * 64 + wv * 16;

  // Q A-fragments (d 0..31 and 32..63), contiguous loads from transposed qt.
  const _Float16* qrow = qt + ((size_t)bidx * NSP + qbase + ln) * CH + h * DHEAD;
  AFrag a0, a1;
  a0.h[0] = *(const v8h*)(qrow + 16 * hl);
  a0.h[1] = *(const v8h*)(qrow + 16 * hl + 8);
  a1.h[0] = *(const v8h*)(qrow + 32 + 16 * hl);
  a1.h[1] = *(const v8h*)(qrow + 32 + 16 * hl + 8);

  v8f o[4] = {};
  float mrow[8], lrow[8];
#pragma unroll
  for (int r = 0; r < 8; ++r) { mrow[r] = -1e30f; lrow[r] = 0.f; }

  const _Float16* kbp = ktb + (size_t)bidx * NSP * CH + h * DHEAD + 16 * hl;
  const _Float16* vr[4];
#pragma unroll
  for (int t = 0; t < 4; ++t)
    vr[t] = vbf + ((size_t)bidx * CH + h * DHEAD + t * 16 + ln) * NSP + 16 * hl;

  for (int kt = 0; kt < 128; ++kt) {
    const int kb = kt * 32;
    // ---- S = (Q . K) / sqrt(dh): two 16x16 tiles over 32 keys ----
    v8f s0 = {}, s1 = {};
    const _Float16* kr0 = kbp + (size_t)(kb + ln) * CH;
    const _Float16* kr1 = kbp + (size_t)(kb + 16 + ln) * CH;
    s0 = wmma_f16f32(a0.v, *(const v16h*)(kr0),      s0);
    s0 = wmma_f16f32(a1.v, *(const v16h*)(kr0 + 32), s0);
    s1 = wmma_f16f32(a0.v, *(const v16h*)(kr1),      s1);
    s1 = wmma_f16f32(a1.v, *(const v16h*)(kr1 + 32), s1);
    s0 = s0 * 0.125f;
    s1 = s1 * 0.125f;

    // Issue V-tile loads now; the softmax VALU chain below hides them.
    v16h bv[4];
#pragma unroll
    for (int t = 0; t < 4; ++t) bv[t] = *(const v16h*)(vr[t] + kb);

    if (kt + 1 < 128) {      // global_prefetch_b8 of the next K/V tiles
      __builtin_prefetch((const void*)(kbp + (size_t)(kb + 32 + ln) * CH), 0, 0);
      __builtin_prefetch((const void*)(vr[0] + kb + 32), 0, 0);
    }

    // ---- online softmax (row stats replicated across 16-lane halves) ----
    float alpha[8];
#pragma unroll
    for (int r = 0; r < 8; ++r) {
      float rm = redmax16(fmaxf(s0[r], s1[r]));
      float mn = fmaxf(mrow[r], rm);
      alpha[r] = expf(mrow[r] - mn);
      mrow[r]  = mn;
    }
#pragma unroll
    for (int r = 0; r < 8; ++r) {
      s0[r] = expf(s0[r] - mrow[r]);
      s1[r] = expf(s1[r] - mrow[r]);
      lrow[r] = lrow[r] * alpha[r] + redsum16(s0[r] + s1[r]);
    }
#pragma unroll
    for (int t = 0; t < 4; ++t)
#pragma unroll
      for (int r = 0; r < 8; ++r) o[t][r] *= alpha[r];

    // ---- P: C-layout -> A-layout via per-wave LDS bounce ----
    wave_lds_fence();                         // prior iteration's reads done
#pragma unroll
    for (int r = 0; r < 8; ++r) {
      pLds[wv][r + 8 * hl][ln]      = (_Float16)s0[r];
      pLds[wv][r + 8 * hl][16 + ln] = (_Float16)s1[r];
    }
    wave_lds_fence();                         // stores visible to wave's loads
    AFrag pa;
    pa.h[0] = *(const v8h*)&pLds[wv][ln][16 * hl];
    pa.h[1] = *(const v8h*)&pLds[wv][ln][16 * hl + 8];

    // ---- O += P * V ----
#pragma unroll
    for (int t = 0; t < 4; ++t)
      o[t] = wmma_f16f32(pa.v, bv[t], o[t]);
  }

  float inv[8];
#pragma unroll
  for (int r = 0; r < 8; ++r) inv[r] = 1.0f / lrow[r];
#pragma unroll
  for (int t = 0; t < 4; ++t) {
    const int ch = h * DHEAD + t * 16 + ln;
#pragma unroll
    for (int r = 0; r < 8; ++r)
      at[((size_t)bidx * NSP + qbase + r + 8 * hl) * CH + ch] = (_Float16)(o[t][r] * inv[r]);
  }
}

// ---------------------------------------------------------------------------
// Kernel 4: proj + bias + residual. Computed as out^T = at[n,c] * proj_w^T,
// so both fragments load contiguously; C tile is (M=spatial, N=oc) so each
// lane stores 8 consecutive f32 (two float4) with fused bias + x residual.
// ---------------------------------------------------------------------------
__global__ void __launch_bounds__(128)
proj_kernel(const _Float16* __restrict__ at, const _Float16* __restrict__ wp16,
            const float* __restrict__ pb, const float* __restrict__ x,
            float* __restrict__ out) {
  const int lane = threadIdx.x & 31;
  const int wv   = threadIdx.x >> 5;
  const int task = blockIdx.x * 4 + wv;        // 2048 tasks
  const int bidx = task >> 10;
  const int rem  = task & 1023;
  const int Sp = (rem >> 2) * 16;              // spatial tile
  const int Ob = (rem & 3) * 64;               // out-channel strip
  const int ln = lane & 15, hl = lane >> 4;

  v8f acc[4] = {};
  const _Float16* arow = at + ((size_t)bidx * NSP + Sp + ln) * CH;
#pragma unroll
  for (int kk = 0; kk < 8; ++kk) {
    const int kb = kk * 32;
    AFrag a;
    a.h[0] = *(const v8h*)(arow + kb + 16 * hl);
    a.h[1] = *(const v8h*)(arow + kb + 16 * hl + 8);
#pragma unroll
    for (int t = 0; t < 4; ++t) {
      v16h bv = *(const v16h*)(wp16 + (size_t)(Ob + t * 16 + ln) * CH + kb + 16 * hl);
      acc[t] = wmma_f16f32(a.v, bv, acc[t]);
    }
  }
#pragma unroll
  for (int t = 0; t < 4; ++t) {
    const int oc = Ob + t * 16 + ln;
    const float bias = pb[oc];
    const size_t base = (size_t)bidx * CH * NSP + (size_t)oc * NSP + Sp + 8 * hl;
    const v4f x0 = *(const v4f*)(x + base);
    const v4f x1 = *(const v4f*)(x + base + 4);
    v4f r0, r1;
#pragma unroll
    for (int j = 0; j < 4; ++j) {
      r0[j] = acc[t][j]     + bias + x0[j];
      r1[j] = acc[t][j + 4] + bias + x1[j];
    }
    *(v4f*)(out + base)     = r0;
    *(v4f*)(out + base + 4) = r1;
  }
}

// ---------------------------------------------------------------------------
extern "C" void kernel_launch(void* const* d_in, const int* in_sizes, int n_in,
                              void* d_out, int out_size, void* d_ws, size_t ws_size,
                              hipStream_t stream) {
  const float* x      = (const float*)d_in[0];
  const float* gn_w   = (const float*)d_in[1];
  const float* gn_b   = (const float*)d_in[2];
  const float* qkv_w  = (const float*)d_in[3];
  const float* qkv_b  = (const float*)d_in[4];
  const float* proj_w = (const float*)d_in[5];
  const float* proj_b = (const float*)d_in[6];
  float* out = (float*)d_out;

  char* wsb = (char*)d_ws;
  _Float16* xnt  = (_Float16*)(wsb);                 // 4 MiB; reused as attn out
  _Float16* qt   = (_Float16*)(wsb + (4u  << 20));   // 4 MiB  q transposed [n][c]
  _Float16* ktb  = (_Float16*)(wsb + (8u  << 20));   // 4 MiB  k transposed [n][c]
  _Float16* vbf  = (_Float16*)(wsb + (12u << 20));   // 4 MiB  v [c][n]
  _Float16* wq16 = (_Float16*)(wsb + (16u << 20));   // 384 KiB
  _Float16* wp16 = (_Float16*)(wsb + (16u << 20) + (size_t)3 * CH * CH * sizeof(_Float16));

  cvt_w_kernel<<<128, 256, 0, stream>>>(qkv_w, proj_w, wq16, wp16);
  gn_kernel<<<BATCH * 8, 256, 0, stream>>>(x, gn_w, gn_b, xnt);
  qkv_kernel<<<1536, 128, 0, stream>>>(wq16, qkv_b, xnt, qt, ktb, vbf);
  attn_kernel<<<BATCH * NHEADS * (NSP / 64), 128, 0, stream>>>(qt, ktb, vbf, xnt);
  proj_kernel<<<512, 128, 0, stream>>>(xnt, wp16, proj_b, x, out);

  (void)in_sizes; (void)n_in; (void)out_size; (void)ws_size;
}
